// LocalChunkScorer_51634096832662
// MI455X (gfx1250) — matrix operand order repaired and checked by reading, not compiled, not run on hardware
//
#include <hip/hip_runtime.h>
#include <hip/hip_bf16.h>
#include <math.h>

// ---------------------------------------------------------------------------
// LocalChunkScorer for MI455X (gfx1250, wave32, WMMA)
//
// scores[b,j,h] = (qp_h . Wk_h^T) @ Xc^T   with Xc = context[start_j : start_j+128]
// Aq[b] = concat_h (qp_h @ Wk_h^T)  : (512 x 1024), precomputed & packed bf16
// s[b,j] = mean_rows 1/sum_l exp(score - rowmax)   (== max of softmax row)
// out    = softmax_j(s), s=0 where len==0
// ---------------------------------------------------------------------------

typedef __attribute__((ext_vector_type(16))) __bf16 v16bf;
typedef __attribute__((ext_vector_type(8)))  float  v8f;

#define Hdim   1024
#define NH     16
#define DH     64
#define Qlen   32
#define Kctx   8192
#define Jn     64
#define Ln     128
#define Mrows  512      // NH * Qlen
#define MT     32       // Mrows/16  M-tiles
#define KT     32       // Hdim/32   K-steps
#define NT     8        // Ln/16     N-tiles

__device__ __forceinline__ unsigned short f2bf(float f) {
    unsigned u = __float_as_uint(f);
    unsigned r = u + 0x7FFFu + ((u >> 16) & 1u);   // round-to-nearest-even
    return (unsigned short)(r >> 16);
}

// --------------------------- K1: qp = qemb @ Wq + bq ------------------------
__global__ void qp_kernel(const float* __restrict__ qemb,
                          const float* __restrict__ Wq,
                          const float* __restrict__ bq,
                          float* __restrict__ qp) {
    int idx = blockIdx.x * 256 + threadIdx.x;      // B*Q*H = 131072
    int i   = idx & (Hdim - 1);
    int row = idx >> 10;                           // b*32+q
    const float* x = qemb + (size_t)row * Hdim;
    float acc = bq[i];
    for (int k = 0; k < Hdim; ++k) acc = fmaf(x[k], Wq[(size_t)k * Hdim + i], acc);
    qp[idx] = acc;
}

// ------------- K2: Aq = qp_h @ Wk_h^T, packed in WMMA-A fragment order ------
// element idx -> ((b*32+mi)*32+kk)*512 + lane*16 + e
// A 16x32 bf16 layout: lane L (m=L&15, g=L>>4), vgpr v=e>>1, pair p=e&1:
//   k = kk*32 + (v<4?0:16) + g*8 + (v&3)*2 + p
__global__ void aq_pack_kernel(const float* __restrict__ qp,
                               const float* __restrict__ Wk,
                               unsigned short* __restrict__ aqPack) {
    size_t idx = (size_t)blockIdx.x * 256 + threadIdx.x;   // 2,097,152 total
    int e   = (int)(idx & 15);
    int L   = (int)((idx >> 4) & 31);
    int kk  = (int)((idx >> 9) & 31);
    int mi  = (int)((idx >> 14) & 31);
    int b   = (int)(idx >> 19);
    int g = L >> 4, mloc = L & 15;
    int v = e >> 1, p = e & 1;
    int k = kk * 32 + ((v < 4) ? 0 : 16) + g * 8 + (v & 3) * 2 + p;
    int M = mi * 16 + mloc;
    int h = M >> 5, q = M & 31;
    const float* qr = qp + ((size_t)(b * Qlen + q)) * Hdim + h * DH;
    const float* wr = Wk + (size_t)k * Hdim + h * DH;
    float acc = 0.f;
    #pragma unroll 8
    for (int d = 0; d < DH; ++d) acc = fmaf(qr[d], wr[d], acc);
    aqPack[idx] = f2bf(acc);
}

// --------------------------- K3: per-(b,j) hot kernel -----------------------
__launch_bounds__(512)
__global__ void chunk_score_kernel(const float* __restrict__ ctx,
                                   const int*   __restrict__ starts,
                                   const int*   __restrict__ lens,
                                   const unsigned short* __restrict__ aqPack,
                                   float* __restrict__ sArr) {
    const int bid  = blockIdx.x;          // b*64 + j
    const int b    = bid >> 6;
    const int tid  = threadIdx.x;
    const int wave = tid >> 5;            // 0..15 (wave32)
    const int lane = tid & 31;

    const int len   = lens[bid];
    const int start = starts[bid];
    if (len <= 0) { if (tid == 0) sArr[bid] = 0.f; return; }

    // Double-buffered B-fragment stage: 2 x [NT][32 lanes][8 u32] = 16 KB
    __shared__ __align__(32) unsigned ldsB[2][NT * 32 * 8];
    __shared__ float wsum[16];

    const float* ctxBase = ctx + (size_t)(b * Kctx + start) * Hdim;

    // B[k][n] = Xc[n][k]; 32x16 bf16 frag: lane L (n=L&15,g=L>>4), vgpr v:
    //   K = (v<4?0:16) + g*8 + (v&3)*2 (+1)
    auto fill = [&](int kk, int buf) {
        const int k0 = kk * 32;
        #pragma unroll
        for (int it = 0; it < (NT * 256) / 512; ++it) {
            int t  = tid + it * 512;
            int ni = t >> 8;
            int L  = (t >> 3) & 31;
            int v  = t & 7;
            int row = ni * 16 + (L & 15);             // l in chunk, 0..127
            int k   = k0 + ((v < 4) ? 0 : 16) + (L >> 4) * 8 + (v & 3) * 2;
            const float2 p2 = *(const float2*)(ctxBase + (size_t)row * Hdim + k);
            ldsB[buf][t] = (unsigned)f2bf(p2.x) | ((unsigned)f2bf(p2.y) << 16);
        }
    };

    const v16bf* aP = (const v16bf*)aqPack;
    const int mi0 = wave * 2, mi1 = mi0 + 1;

    v8f acc0[NT], acc1[NT];
    v8f zero = {0.f, 0.f, 0.f, 0.f, 0.f, 0.f, 0.f, 0.f};
    #pragma unroll
    for (int ni = 0; ni < NT; ++ni) { acc0[ni] = zero; acc1[ni] = zero; }

    fill(0, 0);
    __syncthreads();

    #pragma unroll 1
    for (int kk = 0; kk < KT; ++kk) {
        const int buf = kk & 1;

        // 1) Issue all B-fragment LDS loads first (one clause, single wait,
        //    and DScnt stays orderable against the ds_stores issued below).
        v16bf bfr[NT];
        #pragma unroll
        for (int ni = 0; ni < NT; ++ni)
            bfr[ni] = *(const v16bf*)&ldsB[buf][(ni * 32 + lane) * 8];

        // 2) A fragments for this k-step (L2-resident, coalesced 32B/lane).
        const v16bf a0 = aP[((size_t)(b * MT + mi0) * KT + kk) * 32 + lane];
        const v16bf a1 = aP[((size_t)(b * MT + mi1) * KT + kk) * 32 + lane];

        // 3) Kick off the global->LDS prefetch of the next k-slice; its
        //    ds_stores trail our ds_loads, so WMMAs only need dscnt<=16.
        if (kk + 1 < KT) fill(kk + 1, buf ^ 1);

        // 4) 16 back-to-back WMMAs.
        #pragma unroll
        for (int ni = 0; ni < NT; ++ni) {
            acc0[ni] = __builtin_amdgcn_wmma_f32_16x16x32_bf16(
                false, a0, false, bfr[ni], (short)0, acc0[ni], false, false);
            acc1[ni] = __builtin_amdgcn_wmma_f32_16x16x32_bf16(
                false, a1, false, bfr[ni], (short)0, acc1[ni], false, false);
        }
        __syncthreads();
    }

    // ---- epilogue: per row (M) masked max + sum-exp; attn_max = 1/sumexp.
    // C layout: vgpr r, lanes 0-15: M=r (N=lane), lanes 16-31: M=8+r.
    const int nloc = lane & 15;
    float total = 0.f;
    #pragma unroll
    for (int half = 0; half < 2; ++half) {
        #pragma unroll
        for (int r = 0; r < 8; ++r) {
            float vals[NT];
            float m = -INFINITY;
            #pragma unroll
            for (int ni = 0; ni < NT; ++ni) {
                float sc = (half ? acc1[ni][r] : acc0[ni][r]) * 0.125f; // /sqrt(64)
                vals[ni] = (ni * 16 + nloc < len) ? sc : -INFINITY;
                m = fmaxf(m, vals[ni]);
            }
            #pragma unroll
            for (int off = 1; off < 16; off <<= 1)
                m = fmaxf(m, __shfl_xor(m, off, 32));   // 16-lane group reduce
            float se = 0.f;
            #pragma unroll
            for (int ni = 0; ni < NT; ++ni)
                if (ni * 16 + nloc < len) se += __expf(vals[ni] - m);
            #pragma unroll
            for (int off = 1; off < 16; off <<= 1)
                se += __shfl_xor(se, off, 32);
            total += 1.0f / se;                         // replicated x16 per row
        }
    }
    #pragma unroll
    for (int off = 1; off < 32; off <<= 1) total += __shfl_xor(total, off, 32);
    if (lane == 0) wsum[wave] = total;
    __syncthreads();
    if (tid == 0) {
        float t = 0.f;
        for (int i = 0; i < 16; ++i) t += wsum[i];
        sArr[bid] = t / (float)(Mrows * 16);            // mean over 512 rows
    }
}

// --------------------------- K4: softmax over chunks ------------------------
__global__ void final_softmax_kernel(const float* __restrict__ sArr,
                                     const int*   __restrict__ lens,
                                     float* __restrict__ out) {
    int b = blockIdx.x, j = threadIdx.x;
    float s = sArr[b * Jn + j];
    if (lens[b * Jn + j] <= 0) s = 0.f;
    __shared__ float sh[Jn];
    sh[j] = s;
    __syncthreads();
    float m = -INFINITY;
    for (int i = 0; i < Jn; ++i) m = fmaxf(m, sh[i]);
    float sum = 0.f;
    for (int i = 0; i < Jn; ++i) sum += __expf(sh[i] - m);
    out[b * Jn + j] = __expf(s - m) / sum;
}

// ---------------------------------------------------------------------------
extern "C" void kernel_launch(void* const* d_in, const int* in_sizes, int n_in,
                              void* d_out, int out_size, void* d_ws, size_t ws_size,
                              hipStream_t stream) {
    const float* qemb   = (const float*)d_in[0];   // (4,32,1024)
    const float* ctx    = (const float*)d_in[1];   // (4,8192,1024)
    const int*   starts = (const int*)  d_in[2];   // (4,64)
    const int*   lens   = (const int*)  d_in[3];   // (4,64)
    const float* Wq     = (const float*)d_in[4];   // (1024,1024)
    const float* bq     = (const float*)d_in[5];   // (1024,)
    const float* Wk     = (const float*)d_in[6];   // (1024,1024)
    float* out = (float*)d_out;                    // (4,64)

    char* ws = (char*)d_ws;
    float*          qp     = (float*)ws;                       // 512 KB
    unsigned short* aqPack = (unsigned short*)(ws + (512 << 10));  // 4 MB
    float*          sArr   = (float*)(ws + (512 << 10) + (4 << 20)); // 1 KB

    qp_kernel<<<dim3(512), dim3(256), 0, stream>>>(qemb, Wq, bq, qp);
    aq_pack_kernel<<<dim3(8192), dim3(256), 0, stream>>>(qp, Wk, aqPack);
    chunk_score_kernel<<<dim3(256), dim3(512), 0, stream>>>(ctx, starts, lens,
                                                            aqPack, sArr);
    final_softmax_kernel<<<dim3(4), dim3(Jn), 0, stream>>>(sArr, lens, out);
}
